// Attention_18957985645187
// MI455X (gfx1250) — matrix operand order repaired and checked by reading, not compiled, not run on hardware
//
#include <hip/hip_runtime.h>

#define BATCH   4096
#define INSZ    4096
#define NPART   16
#define DDIM    256
#define NCHUNK  16          // row chunks for BN partial sums
#define LOG2E   1.4426950408889634f

typedef __bf16 bf16;
typedef __attribute__((ext_vector_type(16))) __bf16   v16bf;
typedef __attribute__((ext_vector_type(8)))  float    v8f;
typedef __attribute__((ext_vector_type(4)))  float    f32x4;
typedef __attribute__((ext_vector_type(4)))  unsigned int u32x4;

union ABFrag {
    v16bf v;
    u32x4 q[2];
};

__device__ __forceinline__ bf16 f2bf(float f) { return (bf16)f; }

__device__ __forceinline__ float half_max(float v) {
    // reduce across the 16 lanes of this wave half (xor masks stay within half)
    v = fmaxf(v, __shfl_xor(v, 1, 32));
    v = fmaxf(v, __shfl_xor(v, 2, 32));
    v = fmaxf(v, __shfl_xor(v, 4, 32));
    v = fmaxf(v, __shfl_xor(v, 8, 32));
    return v;
}
__device__ __forceinline__ float half_sum(float v) {
    v += __shfl_xor(v, 1, 32);
    v += __shfl_xor(v, 2, 32);
    v += __shfl_xor(v, 4, 32);
    v += __shfl_xor(v, 8, 32);
    return v;
}

// ---------------------------------------------------------------------------
// Kernel 1a: BN partial sums. Grid (16 col-blocks, 16 row-chunks), 256 thr.
// Deterministic (no atomics): psum[chunk][col], psum2[chunk][col].
// ---------------------------------------------------------------------------
__global__ __launch_bounds__(256) void bn_part_kernel(
    const float* __restrict__ x, float* __restrict__ psum, float* __restrict__ psum2)
{
    int c  = blockIdx.x * 256 + threadIdx.x;         // 0..4095
    int ry = blockIdx.y;                             // 0..15
    int r0 = ry * (BATCH / NCHUNK);
    float s = 0.f, s2 = 0.f;
    for (int r = r0; r < r0 + BATCH / NCHUNK; ++r) {
        float v = x[(size_t)r * INSZ + c];
        s += v; s2 += v * v;
    }
    psum [ry * INSZ + c] = s;
    psum2[ry * INSZ + c] = s2;
}

// ---------------------------------------------------------------------------
// Kernel 1b: reduce partials -> per-feature scale/shift.
//   xn[b,c] = x[b,c]*sc[c] + sh[c]
// ---------------------------------------------------------------------------
__global__ __launch_bounds__(256) void bn_final_kernel(
    const float* __restrict__ psum, const float* __restrict__ psum2,
    const float* __restrict__ gamma, const float* __restrict__ beta,
    float* __restrict__ sc, float* __restrict__ sh)
{
    int c = blockIdx.x * 256 + threadIdx.x;
    float s = 0.f, s2 = 0.f;
    #pragma unroll
    for (int k = 0; k < NCHUNK; ++k) {
        s  += psum [k * INSZ + c];
        s2 += psum2[k * INSZ + c];
    }
    float mean = s * (1.f / BATCH);
    float var  = s2 * (1.f / BATCH) - mean * mean;
    float rstd = rsqrtf(var + 1e-5f);
    float scale = rstd * gamma[c];
    sc[c] = scale;
    sh[c] = beta[c] - mean * scale;
}

// ---------------------------------------------------------------------------
// Kernel 2: convert WQ/WK/WV (each 256x256 f32, row-major [e][d]) to bf16.
// ---------------------------------------------------------------------------
__global__ __launch_bounds__(256) void cvt_w_kernel(
    const float* __restrict__ wq, const float* __restrict__ wk,
    const float* __restrict__ wv, unsigned short* __restrict__ out)
{
    int i = blockIdx.x * 256 + threadIdx.x;          // 0..65535
    bf16* o = (bf16*)out;
    o[i]          = f2bf(wq[i]);
    o[65536 + i]  = f2bf(wk[i]);
    o[131072 + i] = f2bf(wv[i]);
}

// ---------------------------------------------------------------------------
// Per-wave GEMM: dst(16x256) = Xn(16x256) @ W^T(256x256) + bias, then *OSCALE.
//   A frag (M=p, K=d): lane = M row; elems 0-7 -> K=koff..+7, 8-15 -> K=koff+8..+15
//   B frag (K=d, N=e): lane = N col; elem i -> K = koff + i (koff = 0/16 by half)
//   C frag: elem r -> M = r + 8*(lane>=16), N = lane&15
// TR=true stores transposed [e][p] (Q^T/K^T), TR=false stores [p][e] (V).
// ---------------------------------------------------------------------------
template <bool TR>
__device__ __forceinline__ void gemm_qkv(const bf16* __restrict__ W,
                                         const float* __restrict__ bias,
                                         const bf16* sXn, bf16* dst,
                                         int lane, float oscale)
{
    const int nl   = lane & 15;
    const int h    = lane >> 4;
    const int klo  = h * 8;
    const int koff = h * 16;
    for (int nt = 0; nt < 16; ++nt) {
        const int n = nt * 16 + nl;                  // output column e
        float bv = bias[n];
        v8f acc = { bv, bv, bv, bv, bv, bv, bv, bv };
        #pragma unroll
        for (int kk = 0; kk < 8; ++kk) {
            ABFrag a, bfr;
            const bf16* ab = sXn + nl * DDIM + kk * 32;
            a.q[0] = *(const u32x4*)(ab + koff);         // K = koff..koff+7
            a.q[1] = *(const u32x4*)(ab + koff + 8);     // K = koff+8..+15
            const bf16* wb = W + n * DDIM + kk * 32 + koff;
            bfr.q[0] = *(const u32x4*)(wb);              // K = koff..koff+7
            bfr.q[1] = *(const u32x4*)(wb + 8);          // K = koff+8..+15
            acc = __builtin_amdgcn_wmma_f32_16x16x32_bf16(
                false, a.v, false, bfr.v, (short)0, acc, false, false);
        }
        #pragma unroll
        for (int r = 0; r < 8; ++r) {
            int p = r + 8 * h;
            float v = acc[r] * oscale;
            if (TR) dst[n * NPART + p] = f2bf(v);        // [e][p]
            else    dst[p * DDIM  + n] = f2bf(v);        // [p][e]
        }
    }
}

// ---------------------------------------------------------------------------
// Kernel 3: fused normalize + QKV + attention + residual, one batch per wave.
// ---------------------------------------------------------------------------
__global__ __launch_bounds__(32) void attn_kernel(
    const float* __restrict__ x,
    const float* __restrict__ sc, const float* __restrict__ sh,
    const unsigned short* __restrict__ Wqkv,
    const float* __restrict__ bq, const float* __restrict__ bk,
    const float* __restrict__ bv, float* __restrict__ out)
{
    __shared__ alignas(16) bf16  sXn[NPART * DDIM];  // normalized row [p][d]  8KB
    __shared__ alignas(16) bf16  sQT[DDIM * NPART];  // Q^T * (1/16)   [e][p]  8KB
    __shared__ alignas(16) bf16  sKT[DDIM * NPART];  // K^T            [f][p]  8KB
    __shared__ alignas(16) bf16  sV [NPART * DDIM];  // V              [p][f]  8KB
    __shared__ alignas(16) bf16  sP [16 * 32];       // P tile (e x 32f)       1KB
    __shared__ alignas(16) float sProd[INSZ];        // prod flat [e*16+p]    16KB

    const int b    = blockIdx.x;
    const int lane = threadIdx.x;
    const int nl   = lane & 15;
    const int h    = lane >> 4;
    const int klo  = h * 8;
    const int koff = h * 16;
    const float* xr = x + (size_t)b * INSZ;

    // 1) normalize row into LDS (bf16); flat index j == p*256 + d
    for (int j = lane * 4; j < INSZ; j += 128) {
        f32x4 xv = *(const f32x4*)(xr + j);
        f32x4 sv = *(const f32x4*)(sc + j);
        f32x4 hv = *(const f32x4*)(sh + j);
        sXn[j + 0] = f2bf(xv.x * sv.x + hv.x);
        sXn[j + 1] = f2bf(xv.y * sv.y + hv.y);
        sXn[j + 2] = f2bf(xv.z * sv.z + hv.z);
        sXn[j + 3] = f2bf(xv.w * sv.w + hv.w);
    }
    __syncthreads();

    // 2) Q, K, V via WMMA bf16 GEMMs (1/sqrt(d) folded into Q^T)
    const bf16* Wq = (const bf16*)Wqkv;
    gemm_qkv<true >(Wq,          bq, sXn, sQT, lane, 0.0625f);
    gemm_qkv<true >(Wq +  65536, bk, sXn, sKT, lane, 1.0f);
    gemm_qkv<false>(Wq + 131072, bv, sXn, sV , lane, 1.0f);
    __syncthreads();

    // 3) attention: S = Q^T K (K-dim = 16 partitions, padded to 32),
    //    online softmax over f in chunks of 32, P @ V with full K=32
    const u32x4 zf = { 0u, 0u, 0u, 0u };
    for (int et = 0; et < 16; ++et) {
        ABFrag aq;                                   // A[m=e, k=p] = QT[e][p]
        aq.q[0] = h ? zf : *(const u32x4*)(sQT + (et * 16 + nl) * NPART);
        aq.q[1] = h ? zf : *(const u32x4*)(sQT + (et * 16 + nl) * NPART + 8); // K = 16..31 (lanes 16-31) padded zero

        float mr[8], lr[8];
        v8f o = { 0.f, 0.f, 0.f, 0.f, 0.f, 0.f, 0.f, 0.f };
        #pragma unroll
        for (int r = 0; r < 8; ++r) { mr[r] = -3.0e38f; lr[r] = 0.f; }

        for (int fc = 0; fc < 8; ++fc) {             // 32 f-values per step
            // two S tiles: B[k=p, n=f] = KT[f][p]; lanes 16-31 carry K>=16 -> 0
            ABFrag bk0, bk1;
            if (h == 0) {
                const bf16* kb = sKT + (fc * 32 + nl) * NPART;
                bk0.q[0] = *(const u32x4*)(kb);
                bk0.q[1] = *(const u32x4*)(kb + 8);
                bk1.q[0] = *(const u32x4*)(kb + 16 * NPART);
                bk1.q[1] = *(const u32x4*)(kb + 16 * NPART + 8);
            } else {
                bk0.q[0] = zf; bk0.q[1] = zf;
                bk1.q[0] = zf; bk1.q[1] = zf;
            }
            v8f s0 = { 0.f, 0.f, 0.f, 0.f, 0.f, 0.f, 0.f, 0.f };
            v8f s1 = { 0.f, 0.f, 0.f, 0.f, 0.f, 0.f, 0.f, 0.f };
            s0 = __builtin_amdgcn_wmma_f32_16x16x32_bf16(
                false, aq.v, false, bk0.v, (short)0, s0, false, false);
            s1 = __builtin_amdgcn_wmma_f32_16x16x32_bf16(
                false, aq.v, false, bk1.v, (short)0, s1, false, false);

            // online softmax per row (row M = r + 8h lives in this half)
            #pragma unroll
            for (int r = 0; r < 8; ++r) {
                float rm = fmaxf(half_max(s0[r]), half_max(s1[r]));
                float mn = fmaxf(mr[r], rm);
                float sf  = exp2f((mr[r] - mn) * LOG2E);
                float pv0 = exp2f((s0[r] - mn) * LOG2E);
                float pv1 = exp2f((s1[r] - mn) * LOG2E);
                mr[r] = mn;
                o[r] *= sf;
                lr[r] = lr[r] * sf + half_sum(pv0 + pv1);
                sP[(r + 8 * h) * 32 + nl]      = f2bf(pv0);
                sP[(r + 8 * h) * 32 + 16 + nl] = f2bf(pv1);
            }
            __syncthreads();                          // order P store -> load

            // A[m=e, k=f0..31] = P[e][f]: fully populated fragment
            ABFrag ap;
            ap.q[0] = *(const u32x4*)(sP + nl * 32 + koff);
            ap.q[1] = *(const u32x4*)(sP + nl * 32 + koff + 8);

            // B[k=f, n=p] = V[p][f]: fully populated, no divergence
            ABFrag bvf;
            const bf16* vb = sV + nl * DDIM + fc * 32 + koff;
            bvf.q[0] = *(const u32x4*)(vb);
            bvf.q[1] = *(const u32x4*)(vb + 8);

            o = __builtin_amdgcn_wmma_f32_16x16x32_bf16(
                false, ap.v, false, bvf.v, (short)0, o, false, false);
            __syncthreads();
        }

        // stage prod in LDS for coalesced writeout
        #pragma unroll
        for (int r = 0; r < 8; ++r) {
            int e = et * 16 + r + 8 * h;
            sProd[e * NPART + nl] = o[r] / lr[r];
        }
    }
    __syncthreads();

    // 4) coalesced residual writeout: out[b, :] = prod + x[b, :]
    for (int j = lane * 4; j < INSZ; j += 128) {
        f32x4 pv = *(const f32x4*)(sProd + j);
        f32x4 xv = *(const f32x4*)(xr + j);
        *(f32x4*)(out + (size_t)b * INSZ + j) = pv + xv;
    }
}

// ---------------------------------------------------------------------------
extern "C" void kernel_launch(void* const* d_in, const int* in_sizes, int n_in,
                              void* d_out, int out_size, void* d_ws, size_t ws_size,
                              hipStream_t stream)
{
    const float* x     = (const float*)d_in[0];
    const float* WQ_w  = (const float*)d_in[1];
    const float* WQ_b  = (const float*)d_in[2];
    const float* WK_w  = (const float*)d_in[3];
    const float* WK_b  = (const float*)d_in[4];
    const float* WV_w  = (const float*)d_in[5];
    const float* WV_b  = (const float*)d_in[6];
    const float* gamma = (const float*)d_in[7];
    const float* beta  = (const float*)d_in[8];
    float* out = (float*)d_out;

    // ws layout: sc[4096] | sh[4096] | psum[16*4096] | psum2[16*4096] (f32)
    //            | Wqkv 3*65536 bf16
    float* sc    = (float*)d_ws;
    float* sh    = sc + INSZ;
    float* psum  = sh + INSZ;
    float* psum2 = psum + NCHUNK * INSZ;
    unsigned short* Wqkv = (unsigned short*)(psum2 + NCHUNK * INSZ);

    bn_part_kernel<<<dim3(INSZ / 256, NCHUNK), 256, 0, stream>>>(x, psum, psum2);
    bn_final_kernel<<<INSZ / 256, 256, 0, stream>>>(psum, psum2, gamma, beta, sc, sh);
    cvt_w_kernel<<<65536 / 256, 256, 0, stream>>>(WQ_w, WK_w, WV_w, Wqkv);
    attn_kernel<<<BATCH, 32, 0, stream>>>(x, sc, sh, Wqkv,
                                          WQ_b, WK_b, WV_b, out);
}
